// DRP_11252814315652
// MI455X (gfx1250) — matrix operand order repaired and checked by reading, not compiled, not run on hardware
//
#include <hip/hip_runtime.h>
#include <hip/hip_bf16.h>
#include <stddef.h>

// ---------------------------------------------------------------------------
// PointConv-style network on MI455X (gfx1250).
// Dense math: v_wmma_f32_16x16x32_bf16, 128x64 block tile, 32x32 per wave,
// double-buffered LDS (one barrier per K-step).
// ---------------------------------------------------------------------------

typedef __attribute__((ext_vector_type(16))) __bf16 v16bf;
typedef __attribute__((ext_vector_type(8)))  __bf16 v8bf;
typedef __attribute__((ext_vector_type(8)))  float  v8f;

#define LDSX 40   // halves per LDS row (80B row pitch: all b128 accesses 16B-aligned)

static __host__ __device__ inline int pad32i(int k) { return (k + 31) & ~31; }

// ---------------------------------------------------------------------------
// WMMA GEMM: Y[R,C] = X[R,Kp] * W[C,Kp]^T   (bf16 x bf16 -> f32 accum)
// grid = (C/64, R/128), block = 256 (8 wave32).
// Waves: 4 M-tiles x 2 N-tiles; each wave computes 32x32 = 4 WMMA tiles.
// ---------------------------------------------------------------------------
__global__ __launch_bounds__(256) void gemm_bf16_wmma(
    const __bf16* __restrict__ X, const __bf16* __restrict__ W,
    float* __restrict__ Y, int R, int Kp, int C)
{
  __shared__ __bf16 lX[2][128 * LDSX];
  __shared__ __bf16 lW[2][64 * LDSX];

  const int tid   = threadIdx.x;
  const int lane  = tid & 31;
  const int wave  = tid >> 5;
  const int waveM = wave & 3;        // 0..3 -> 32-row strip
  const int waveN = wave >> 2;       // 0..1 -> 32-col strip
  const int l16   = lane & 15;
  const int half  = (lane >> 4) & 1; // lanes 0-15 vs 16-31

  const int rowBase = blockIdx.y * 128;
  const int colBase = blockIdx.x * 64;

  v8f acc00 = {}, acc01 = {}, acc10 = {}, acc11 = {};

  // cooperative stage of one K-slab (X: 128x32, W: 64x32) into LDS buffer `b`
  auto stage = [&](int k0, int b) {
    int r = tid >> 1;                 // 0..127
    int c = (tid & 1) * 16;           // 0 or 16
    const __bf16* src = X + (size_t)(rowBase + r) * Kp + k0 + c;
    __bf16* dst = &lX[b][r * LDSX + c];
    *(v8bf*)dst       = *(const v8bf*)src;
    *(v8bf*)(dst + 8) = *(const v8bf*)(src + 8);
    int rw = tid >> 2;                // 0..63
    int cw = (tid & 3) * 8;           // 0,8,16,24
    *(v8bf*)&lW[b][rw * LDSX + cw] =
        *(const v8bf*)(W + (size_t)(colBase + rw) * Kp + k0 + cw);
  };

  stage(0, 0);
  __syncthreads();

  int buf = 0;
  for (int k0 = 0; k0 < Kp; k0 += 32) {
    if (k0 + 32 < Kp) {
      stage(k0 + 32, buf ^ 1);        // overlaps with WMMA on current buffer
      __builtin_prefetch(X + (size_t)(rowBase + (tid >> 1)) * Kp + k0 + 64, 0, 3);
    }

    // --- A fragments (ISA 16-bit A 16x32 layout), rows waveM*32 + {0,16} ---
    const int kb = half * 8;
    const int mA = waveM * 32 + l16;
    v8bf a0lo = *(const v8bf*)&lX[buf][(mA +  0) * LDSX + kb];
    v8bf a0hi = *(const v8bf*)&lX[buf][(mA +  0) * LDSX + kb + 16];
    v8bf a1lo = *(const v8bf*)&lX[buf][(mA + 16) * LDSX + kb];
    v8bf a1hi = *(const v8bf*)&lX[buf][(mA + 16) * LDSX + kb + 16];
    v16bf a0 = __builtin_shufflevector(a0lo, a0hi, 0,1,2,3,4,5,6,7,8,9,10,11,12,13,14,15);
    v16bf a1 = __builtin_shufflevector(a1lo, a1hi, 0,1,2,3,4,5,6,7,8,9,10,11,12,13,14,15);

    // --- B fragments (ISA 16-bit B 32x16 layout), cols waveN*32 + {0,16} ---
    const int ks = half * 8;
    const int nB = waveN * 32 + l16;
    v8bf b0lo = *(const v8bf*)&lW[buf][(nB +  0) * LDSX + ks];
    v8bf b0hi = *(const v8bf*)&lW[buf][(nB +  0) * LDSX + ks + 16];
    v8bf b1lo = *(const v8bf*)&lW[buf][(nB + 16) * LDSX + ks];
    v8bf b1hi = *(const v8bf*)&lW[buf][(nB + 16) * LDSX + ks + 16];
    v16bf b0 = __builtin_shufflevector(b0lo, b0hi, 0,1,2,3,4,5,6,7,8,9,10,11,12,13,14,15);
    v16bf b1 = __builtin_shufflevector(b1lo, b1hi, 0,1,2,3,4,5,6,7,8,9,10,11,12,13,14,15);

    acc00 = __builtin_amdgcn_wmma_f32_16x16x32_bf16(false, a0, false, b0, (short)0, acc00, false, false);
    acc01 = __builtin_amdgcn_wmma_f32_16x16x32_bf16(false, a0, false, b1, (short)0, acc01, false, false);
    acc10 = __builtin_amdgcn_wmma_f32_16x16x32_bf16(false, a1, false, b0, (short)0, acc10, false, false);
    acc11 = __builtin_amdgcn_wmma_f32_16x16x32_bf16(false, a1, false, b1, (short)0, acc11, false, false);

    __syncthreads();                  // readers of `buf` done before it is restaged
    buf ^= 1;
  }

  // --- store per 32-bit C/D layout: VGPR r -> (M = r + half*8, N = l16) ---
  const int n0 = colBase + waveN * 32 + l16;
  const int m0 = rowBase + waveM * 32 + half * 8;
#pragma unroll
  for (int r = 0; r < 8; ++r) {
    float* yr0 = Y + (size_t)(m0 + r) * C;
    float* yr1 = Y + (size_t)(m0 + 16 + r) * C;
    yr0[n0]      = acc00[r];
    yr0[n0 + 16] = acc01[r];
    yr1[n0]      = acc10[r];
    yr1[n0 + 16] = acc11[r];
  }
}

// ---------------------------------------------------------------------------
// Farthest point sampling: one block per batch, serial npoint iterations.
// ---------------------------------------------------------------------------
__global__ void fps_kernel(const float* __restrict__ xyz, int N, int npoint,
                           int* __restrict__ out, float* __restrict__ dmin)
{
  const int b  = blockIdx.x;
  const float* p = xyz + (size_t)b * N * 3;
  float* d = dmin + (size_t)b * N;
  int*   o = out  + (size_t)b * npoint;
  __shared__ float sval[256];
  __shared__ int   sidx[256];
  __shared__ float cen[3];
  const int tid = threadIdx.x;

  for (int i = tid; i < N; i += 256) d[i] = 1e10f;
  __syncthreads();

  int far = 0;
  for (int it = 0; it < npoint; ++it) {
    if (tid == 0) {
      o[it] = far;
      cen[0] = p[far * 3 + 0]; cen[1] = p[far * 3 + 1]; cen[2] = p[far * 3 + 2];
    }
    __syncthreads();
    float cx = cen[0], cy = cen[1], cz = cen[2];
    float best = -1.0f; int bi = 0;
    for (int i = tid; i < N; i += 256) {
      float dx = p[i*3] - cx, dy = p[i*3+1] - cy, dz = p[i*3+2] - cz;
      float dd = dx*dx + dy*dy + dz*dz;
      float dm = d[i]; dm = dm < dd ? dm : dd; d[i] = dm;
      if (dm > best) { best = dm; bi = i; }
    }
    sval[tid] = best; sidx[tid] = bi;
    __syncthreads();
    for (int s = 128; s > 0; s >>= 1) {
      if (tid < s) {
        if (sval[tid+s] > sval[tid] ||
            (sval[tid+s] == sval[tid] && sidx[tid+s] < sidx[tid])) {
          sval[tid] = sval[tid+s]; sidx[tid] = sidx[tid+s];
        }
      }
      __syncthreads();
    }
    far = sidx[0];
    __syncthreads();
  }
}

__global__ void gather3_kernel(const float* __restrict__ xyz, const int* __restrict__ fi,
                               float* __restrict__ out, int N, int M, int total)
{
  int g = blockIdx.x * blockDim.x + threadIdx.x;
  if (g >= total) return;
  int b = g / M;
  int id = fi[g];
  const float* p = xyz + ((size_t)b * N + id) * 3;
  out[(size_t)g*3+0] = p[0]; out[(size_t)g*3+1] = p[1]; out[(size_t)g*3+2] = p[2];
}

// Ball query: first `ns` support indices (ascending) within radius; pad w/ first.
__global__ void ball_query_kernel(const float* __restrict__ support,
                                  const float* __restrict__ query,
                                  int N, int M, float r2, int ns,
                                  int* __restrict__ idx, int total)
{
  int g = blockIdx.x * blockDim.x + threadIdx.x;
  if (g >= total) return;
  int b = g / M;
  const float* s = support + (size_t)b * N * 3;
  float qx = query[(size_t)g*3], qy = query[(size_t)g*3+1], qz = query[(size_t)g*3+2];
  int* o = idx + (size_t)g * ns;
  int cnt = 0, first = 0;
  for (int n = 0; n < N && cnt < ns; ++n) {
    float dx = s[n*3] - qx, dy = s[n*3+1] - qy, dz = s[n*3+2] - qz;
    if (dx*dx + dy*dy + dz*dz < r2) { if (cnt == 0) first = n; o[cnt++] = n; }
  }
  for (int k = cnt; k < ns; ++k) o[k] = first;
}

// Build grouped GEMM input: [(xyz[idx]-center)*scale, feat[idx], 0-pad] -> bf16.
__global__ void group_pack_kernel(const float* __restrict__ xyz,
                                  const float* __restrict__ centers,
                                  const int* __restrict__ idx,
                                  const float* __restrict__ feat,
                                  int N, int M, int ns, int C,
                                  float scale, int Kp, __bf16* __restrict__ Xp)
{
  int row = blockIdx.x;                  // (b*M+m)*ns + j
  int j   = row % ns;
  int g   = row / ns;
  int b   = g / M;
  int id  = idx[(size_t)g * ns + j];
  const float* pt  = xyz + ((size_t)b * N + id) * 3;
  const float* cen = centers + (size_t)g * 3;
  __bf16* out = Xp + (size_t)row * Kp;
  for (int c = threadIdx.x; c < Kp; c += blockDim.x) {
    float v = 0.0f;
    if (c < 3)           v = (pt[c] - cen[c]) * scale;
    else if (c < 3 + C)  v = feat[((size_t)b * N + id) * C + (c - 3)];
    out[c] = (__bf16)v;
  }
}

__global__ void pack_bf16_kernel(const float* __restrict__ X, __bf16* __restrict__ Xp,
                                 int C, int Kp)
{
  int row = blockIdx.x;
  const float* src = X + (size_t)row * C;
  __bf16* dst = Xp + (size_t)row * Kp;
  for (int c = threadIdx.x; c < Kp; c += blockDim.x)
    dst[c] = (__bf16)(c < C ? src[c] : 0.0f);
}

__global__ void pack_w_kernel(const float* __restrict__ w, __bf16* __restrict__ wp,
                              int K, int Kp)
{
  int o = blockIdx.x;
  for (int c = threadIdx.x; c < Kp; c += blockDim.x)
    wp[(size_t)o * Kp + c] = (__bf16)(c < K ? w[(size_t)o * K + c] : 0.0f);
}

// Per-channel mean / rsqrt(var+eps). Block owns 64 adjacent channels so each
// row-step is a contiguous 256B read (coalesced); 4 row-phases combined in LDS.
__global__ void bn_stats_kernel(const float* __restrict__ Y, int R, int C,
                                float* __restrict__ mean, float* __restrict__ rstd)
{
  const int tx = threadIdx.x & 63;
  const int ty = threadIdx.x >> 6;     // 0..3
  const int c  = blockIdx.x * 64 + tx;
  float s = 0.f, ss = 0.f;
  for (int r = ty; r < R; r += 4) {
    float v = Y[(size_t)r * C + c];
    s += v; ss += v * v;
  }
  __shared__ float rs[256], rq[256];
  rs[threadIdx.x] = s; rq[threadIdx.x] = ss;
  __syncthreads();
  if (ty == 0) {
    for (int j = 1; j < 4; ++j) { s += rs[tx + j*64]; ss += rq[tx + j*64]; }
    float m = s / (float)R;
    float v = ss / (float)R - m * m;
    mean[c] = m;
    rstd[c] = rsqrtf(v + 1e-5f);
  }
}

__global__ void bn_act_kernel(const float* __restrict__ Y,
                              const float* __restrict__ mean, const float* __restrict__ rstd,
                              const float* __restrict__ gamma, const float* __restrict__ beta,
                              float* __restrict__ out, size_t total, int C, int relu)
{
  size_t i = blockIdx.x * (size_t)blockDim.x + threadIdx.x;
  if (i >= total) return;
  int c = (int)(i % (size_t)C);
  float v = gamma[c] * (Y[i] - mean[c]) * rstd[c] + beta[c];
  if (relu) v = v > 0.f ? v : 0.f;
  out[i] = v;
}

__global__ void maxpool_kernel(const float* __restrict__ X, float* __restrict__ out,
                               int ns, int C)
{
  int g = blockIdx.x;
  for (int c = threadIdx.x; c < C; c += blockDim.x) {
    float m = -3.4e38f;
    for (int j = 0; j < ns; ++j) {
      float v = X[((size_t)g * ns + j) * C + c];
      m = v > m ? v : m;
    }
    out[(size_t)g * C + c] = m;
  }
}

__global__ void add_relu_kernel(const float* __restrict__ a, const float* __restrict__ b,
                                float* __restrict__ o, size_t n)
{
  size_t i = blockIdx.x * (size_t)blockDim.x + threadIdx.x;
  if (i >= n) return;
  float v = a[i] + b[i];
  o[i] = v > 0.f ? v : 0.f;
}

// 3-NN inverse-distance interpolation + concat -> bf16 packed GEMM input.
__global__ void interp_pack_kernel(const float* __restrict__ xq, const float* __restrict__ xs,
                                   const float* __restrict__ fq, const float* __restrict__ fs,
                                   int M1, int M2, int C1, int C2, int Kp,
                                   __bf16* __restrict__ Xp)
{
  int g = blockIdx.x;              // b*M1 + m
  int b = g / M1;
  __shared__ int   nn[3];
  __shared__ float wt[3];
  if (threadIdx.x == 0) {
    float px = xq[(size_t)g*3], py = xq[(size_t)g*3+1], pz = xq[(size_t)g*3+2];
    const float* s = xs + (size_t)b * M2 * 3;
    float bd0 = 1e30f, bd1 = 1e30f, bd2 = 1e30f;
    int   bi0 = 0, bi1 = 0, bi2 = 0;
    for (int n = 0; n < M2; ++n) {
      float dx = s[n*3] - px, dy = s[n*3+1] - py, dz = s[n*3+2] - pz;
      float d = dx*dx + dy*dy + dz*dz;
      if (d < bd0)      { bd2=bd1; bi2=bi1; bd1=bd0; bi1=bi0; bd0=d; bi0=n; }
      else if (d < bd1) { bd2=bd1; bi2=bi1; bd1=d;   bi1=n; }
      else if (d < bd2) { bd2=d;   bi2=n; }
    }
    float w0 = 1.f/(bd0+1e-8f), w1 = 1.f/(bd1+1e-8f), w2 = 1.f/(bd2+1e-8f);
    float ws = w0 + w1 + w2;
    nn[0]=bi0; nn[1]=bi1; nn[2]=bi2;
    wt[0]=w0/ws; wt[1]=w1/ws; wt[2]=w2/ws;
  }
  __syncthreads();
  __bf16* out = Xp + (size_t)g * Kp;
  const float* fb = fs + (size_t)b * M2 * C2;
  for (int c = threadIdx.x; c < Kp; c += blockDim.x) {
    float v = 0.f;
    if (c < C2)
      v = wt[0]*fb[(size_t)nn[0]*C2+c] + wt[1]*fb[(size_t)nn[1]*C2+c] + wt[2]*fb[(size_t)nn[2]*C2+c];
    else if (c < C2 + C1)
      v = fq[(size_t)g * C1 + (c - C2)];
    out[c] = (__bf16)v;
  }
}

__global__ void transpose_out_kernel(const float* __restrict__ f, float* __restrict__ out,
                                     int M, int C, size_t total)
{
  size_t i = blockIdx.x * (size_t)blockDim.x + threadIdx.x;
  if (i >= total) return;
  int c = (int)(i % (size_t)C);
  size_t bm = i / (size_t)C;
  int m = (int)(bm % (size_t)M);
  int b = (int)(bm / (size_t)M);
  out[((size_t)b * C + c) * M + m] = f[i];
}

__global__ void copy_kernel(const float* __restrict__ s, float* __restrict__ d, size_t n)
{
  size_t i = blockIdx.x * (size_t)blockDim.x + threadIdx.x;
  if (i < n) d[i] = s[i];
}

// ---------------------------------------------------------------------------
// Host orchestration
// ---------------------------------------------------------------------------
namespace {
struct Arena {
  char* p; size_t off; size_t cap;
  void* get(size_t bytes) {
    size_t a = (off + 255) & ~(size_t)255;
    off = a + bytes;
    return p + a;
  }
};
}

extern "C" void kernel_launch(void* const* d_in, const int* in_sizes, int n_in,
                              void* d_out, int out_size, void* d_ws, size_t ws_size,
                              hipStream_t stream) {
  (void)in_sizes; (void)n_in; (void)out_size;
  const int B = 2, N0 = 16384;
  const float* pc = (const float*)d_in[0];

  // Param leaf bases (setup_inputs dict insertion order; cb = {w, gamma, beta}).
  const int SA1 = 1,  INV1 = 10,  SA2 = 37,  INV2 = 46;
  const int SA3 = 100, INV3 = 109, SA4 = 136, INV4 = 145;
  const int FP1 = 172, FP2 = 178;

  Arena A{(char*)d_ws, 0, ws_size};

  // Persistent per-level buffers.
  float* x1 = (float*)A.get((size_t)B*2048*3*4);
  float* f1 = (float*)A.get((size_t)B*2048*128*4);
  float* x2 = (float*)A.get((size_t)B*1024*3*4);
  float* f2 = (float*)A.get((size_t)B*1024*256*4);
  float* x3 = (float*)A.get((size_t)B*512*3*4);
  float* f3 = (float*)A.get((size_t)B*512*256*4);
  float* x4 = (float*)A.get((size_t)B*256*3*4);
  float* f4 = (float*)A.get((size_t)B*256*256*4);
  float* g3 = (float*)A.get((size_t)B*512*256*4);
  float* g2 = (float*)A.get((size_t)B*1024*256*4);

  // conv block: pack W -> WMMA GEMM -> BN stats -> BN+act
  auto run_conv = [&](const __bf16* Xp, int R, int Kp, int cin, int cout,
                      int leaf, bool relu, float* outAct) {
    size_t mk = A.off;
    __bf16* Wp = (__bf16*)A.get((size_t)cout * Kp * 2);
    pack_w_kernel<<<cout, 64, 0, stream>>>((const float*)d_in[leaf], Wp, cin, Kp);
    float* Yr = (float*)A.get((size_t)R * cout * 4);
    dim3 grid(cout / 64, R / 128);
    gemm_bf16_wmma<<<grid, 256, 0, stream>>>(Xp, Wp, Yr, R, Kp, cout);
    float* mean = (float*)A.get((size_t)cout * 4);
    float* rstd = (float*)A.get((size_t)cout * 4);
    bn_stats_kernel<<<cout / 64, 256, 0, stream>>>(Yr, R, cout, mean, rstd);
    size_t tot = (size_t)R * cout;
    bn_act_kernel<<<(unsigned)((tot + 255) / 256), 256, 0, stream>>>(
        Yr, mean, rstd, (const float*)d_in[leaf + 1], (const float*)d_in[leaf + 2],
        outAct, tot, cout, relu ? 1 : 0);
    A.off = mk;
  };

  auto sa_stage = [&](const float* sxyz, int Nn, const float* feat, int C,
                      int M, float radius, int ns, int base,
                      int c1, int c2, int c3, float* ox, float* of) {
    size_t mk = A.off;
    int totq = B * M;
    int* fi = (int*)A.get((size_t)totq * 4);
    float* dscr = (float*)A.get((size_t)B * Nn * 4);
    fps_kernel<<<B, 256, 0, stream>>>(sxyz, Nn, M, fi, dscr);
    gather3_kernel<<<(totq + 255) / 256, 256, 0, stream>>>(sxyz, fi, ox, Nn, M, totq);
    int* idx = (int*)A.get((size_t)totq * ns * 4);
    ball_query_kernel<<<(totq + 255) / 256, 256, 0, stream>>>(
        sxyz, ox, Nn, M, radius * radius, ns, idx, totq);
    int R = totq * ns;
    int Kp0 = pad32i(3 + C);
    __bf16* X0 = (__bf16*)A.get((size_t)R * Kp0 * 2);
    group_pack_kernel<<<R, 64, 0, stream>>>(sxyz, ox, idx, feat, Nn, M, ns, C,
                                            1.0f / radius, Kp0, X0);
    float* a1 = (float*)A.get((size_t)R * c1 * 4);
    run_conv(X0, R, Kp0, 3 + C, c1, base + 0, true, a1);
    int Kp1 = pad32i(c1);
    __bf16* X1 = (__bf16*)A.get((size_t)R * Kp1 * 2);
    pack_bf16_kernel<<<R, 64, 0, stream>>>(a1, X1, c1, Kp1);
    float* a2 = (float*)A.get((size_t)R * c2 * 4);
    run_conv(X1, R, Kp1, c1, c2, base + 3, true, a2);
    int Kp2 = pad32i(c2);
    __bf16* X2 = (__bf16*)A.get((size_t)R * Kp2 * 2);
    pack_bf16_kernel<<<R, 64, 0, stream>>>(a2, X2, c2, Kp2);
    float* a3 = (float*)A.get((size_t)R * c3 * 4);
    run_conv(X2, R, Kp2, c2, c3, base + 6, true, a3);
    maxpool_kernel<<<totq, 128, 0, stream>>>(a3, of, ns, c3);
    A.off = mk;
  };

  auto inv_stage = [&](const float* x, float* f, int M, int C,
                       float radius, int ns, int base) {
    size_t mk = A.off;
    int totq = B * M;
    int* idx = (int*)A.get((size_t)totq * ns * 4);
    ball_query_kernel<<<(totq + 255) / 256, 256, 0, stream>>>(
        x, x, M, M, radius * radius, ns, idx, totq);
    int R = totq * ns;
    int Kp0 = pad32i(3 + C);
    __bf16* X0 = (__bf16*)A.get((size_t)R * Kp0 * 2);
    group_pack_kernel<<<R, 64, 0, stream>>>(x, x, idx, f, M, M, ns, C, 1.0f, Kp0, X0);
    float* aL = (float*)A.get((size_t)R * C * 4);
    run_conv(X0, R, Kp0, 3 + C, C, base + 0, true, aL);       // la
    float* h = (float*)A.get((size_t)totq * C * 4);
    maxpool_kernel<<<totq, 128, 0, stream>>>(aL, h, ns, C);
    int KpC = pad32i(C);
    __bf16* Xh = (__bf16*)A.get((size_t)totq * KpC * 2);
    pack_bf16_kernel<<<totq, 64, 0, stream>>>(h, Xh, C, KpC);
    float* h1 = (float*)A.get((size_t)totq * 4 * C * 4);
    run_conv(Xh, totq, KpC, C, 4 * C, base + 3, true, h1);    // pw1
    int Kp4 = pad32i(4 * C);
    __bf16* Xh1 = (__bf16*)A.get((size_t)totq * Kp4 * 2);
    pack_bf16_kernel<<<totq, 64, 0, stream>>>(h1, Xh1, 4 * C, Kp4);
    float* h2 = (float*)A.get((size_t)totq * C * 4);
    run_conv(Xh1, totq, Kp4, 4 * C, C, base + 6, false, h2);  // pw2 (no act)
    size_t tot = (size_t)totq * C;
    add_relu_kernel<<<(unsigned)((tot + 255) / 256), 256, 0, stream>>>(h2, f, f, tot);
    A.off = mk;
  };

  auto fp_stage = [&](const float* xq, const float* xs, const float* fq, const float* fs,
                      int M1, int M2, int C1, int C2, int base, float* out) {
    size_t mk = A.off;
    int totq = B * M1;
    int Kp = pad32i(C1 + C2);
    __bf16* X = (__bf16*)A.get((size_t)totq * Kp * 2);
    interp_pack_kernel<<<totq, 128, 0, stream>>>(xq, xs, fq, fs, M1, M2, C1, C2, Kp, X);
    float* a1 = (float*)A.get((size_t)totq * 256 * 4);
    run_conv(X, totq, Kp, C1 + C2, 256, base + 0, true, a1);
    int Kp1 = pad32i(256);
    __bf16* X1 = (__bf16*)A.get((size_t)totq * Kp1 * 2);
    pack_bf16_kernel<<<totq, 64, 0, stream>>>(a1, X1, 256, Kp1);
    run_conv(X1, totq, Kp1, 256, 256, base + 3, true, out);
    A.off = mk;
  };

  // ---- full pipeline ----
  sa_stage(pc, N0, nullptr, 0, 2048, 0.04f, 64, SA1, 64, 64, 128, x1, f1);
  for (int j = 0; j < 3; ++j) inv_stage(x1, f1, 2048, 128, 0.08f, 64, INV1 + j * 9);

  sa_stage(x1, 2048, f1, 128, 1024, 0.10f, 32, SA2, 128, 128, 256, x2, f2);
  for (int j = 0; j < 6; ++j) inv_stage(x2, f2, 1024, 256, 0.20f, 32, INV2 + j * 9);

  sa_stage(x2, 1024, f2, 256, 512, 0.20f, 16, SA3, 128, 128, 256, x3, f3);
  for (int j = 0; j < 3; ++j) inv_stage(x3, f3, 512, 256, 0.40f, 16, INV3 + j * 9);

  sa_stage(x3, 512, f3, 256, 256, 0.30f, 16, SA4, 128, 128, 256, x4, f4);
  for (int j = 0; j < 3; ++j) inv_stage(x4, f4, 256, 256, 0.60f, 16, INV4 + j * 9);

  fp_stage(x3, x4, f3, f4, 512, 256, 256, 256, FP1, g3);
  fp_stage(x2, x3, f2, g3, 1024, 512, 256, 256, FP2, g2);

  // ---- outputs: (B,256,1024) transposed features, then x2 (B,1024,3) ----
  size_t tot = (size_t)B * 1024 * 256;
  transpose_out_kernel<<<(unsigned)((tot + 255) / 256), 256, 0, stream>>>(
      g2, (float*)d_out, 1024, 256, tot);
  size_t nx = (size_t)B * 1024 * 3;
  copy_kernel<<<(unsigned)((nx + 255) / 256), 256, 0, stream>>>(
      x2, (float*)d_out + tot, nx);
}
